// GCN_75565654606208
// MI455X (gfx1250) — compile-verified, matrix operand order checked by reading, not run on hardware
//
#include <hip/hip_runtime.h>
#include <hip/hip_bf16.h>
#include <math.h>

#define N_NODES 100000
#define N_EDGES 1600000
#define IN_CH   256
#define HID_CH  128
#define OUT_CH  64

typedef __attribute__((ext_vector_type(16))) __bf16 v16bf;
typedef __attribute__((ext_vector_type(8)))  __bf16 v8bf;
typedef __attribute__((ext_vector_type(8)))  float  v8f;
typedef __attribute__((ext_vector_type(4)))  int    v4i;

#define AS1 __attribute__((address_space(1)))
#define AS3 __attribute__((address_space(3)))

#if defined(__has_builtin)
#if __has_builtin(__builtin_amdgcn_global_load_async_to_lds_b128) && \
    __has_builtin(__builtin_amdgcn_s_wait_asynccnt)
#define USE_ASYNC_LDS 1
#endif
#endif
#ifndef USE_ASYNC_LDS
#define USE_ASYNC_LDS 0
#endif

// Load a 16-bit A/B fragment (ISA 7.12.2 layout): per lane two contiguous
// 8 x bf16 (16B) runs at K = kh..kh+7 and K = kh+16..kh+23.
__device__ __forceinline__ v16bf load_frag(const __bf16* p) {
  v8bf lo = *(const v8bf*)(p);
  v8bf hi = *(const v8bf*)(p + 16);
  v16bf f;
#pragma unroll
  for (int i = 0; i < 8; ++i) { f[i] = lo[i]; f[8 + i] = hi[i]; }
  return f;
}

// ---------------------------------------------------------------------------
// GEMM1: h1[N,128] = x[N,256] @ W1[256,128], bf16 hi/lo split (near-fp32).
// Block: 256 thr = 8 waves, 64 rows x 64 cols (grid.y = 2 column blocks).
// Wave w -> col tile (w&3)*16, row-tile pair ((w>>2)*2). K = 256 = 8 x 32.
// acc += Ahi*Blo + Alo*Bhi + Ahi*Bhi   (fp32 accumulate)
// ---------------------------------------------------------------------------
__global__ __launch_bounds__(256)
void gemm1_x_w1(const float* __restrict__ x, const float* __restrict__ W1,
                float* __restrict__ h1) {
  __shared__ __bf16 xs_hi[64][IN_CH + 8];
  __shared__ __bf16 xs_lo[64][IN_CH + 8];
  __shared__ __bf16 wt_hi[64][IN_CH + 8];   // W1^T slab, 64 output cols
  __shared__ __bf16 wt_lo[64][IN_CH + 8];

  const int tid = threadIdx.x;
  const int r0  = blockIdx.x * 64;
  const int nb  = blockIdx.y * 64;

  // Stage W1^T hi/lo (W1 is [256][128] row-major); coalesced 64-float runs.
  for (int i = tid; i < 64 * IN_CH; i += 256) {
    int k = i >> 6;              // 0..255
    int n = i & 63;
    float v = W1[(size_t)k * HID_CH + nb + n];
    __bf16 h = (__bf16)v;
    wt_hi[n][k] = h;
    wt_lo[n][k] = (__bf16)(v - (float)h);
  }
  // Stage x tile hi/lo
  for (int i = tid; i < 64 * IN_CH; i += 256) {
    int r = i >> 8;
    int c = i & 255;
    float v = (r0 + r < N_NODES) ? x[(size_t)(r0 + r) * IN_CH + c] : 0.0f;
    __bf16 h = (__bf16)v;
    xs_hi[r][c] = h;
    xs_lo[r][c] = (__bf16)(v - (float)h);
  }
  __syncthreads();

  const int wave = tid >> 5, lane = tid & 31;
  const int m  = lane & 15;
  const int kh = (lane >> 4) << 3;
  const int n0 = (wave & 3) << 4;
  const int tb = (wave >> 2) << 1;

  v16bf bhi[8];
#pragma unroll
  for (int kk = 0; kk < 8; ++kk)
    bhi[kk] = load_frag(&wt_hi[n0 + m][kk * 32 + kh]);

  const bool full = (r0 + 64 <= N_NODES);   // block-uniform fast path

#pragma unroll
  for (int tt = 0; tt < 2; ++tt) {
    const int t = tb + tt;
    v8f acc = {};
#pragma unroll
    for (int kk = 0; kk < 8; ++kk) {
      v16bf ahi = load_frag(&xs_hi[t * 16 + m][kk * 32 + kh]);
      v16bf alo = load_frag(&xs_lo[t * 16 + m][kk * 32 + kh]);
      v16bf blo = load_frag(&wt_lo[n0 + m][kk * 32 + kh]);
      acc = __builtin_amdgcn_wmma_f32_16x16x32_bf16(false, ahi, false, blo,
                                                    (short)0, acc, false, false);
      acc = __builtin_amdgcn_wmma_f32_16x16x32_bf16(false, alo, false, bhi[kk],
                                                    (short)0, acc, false, false);
      acc = __builtin_amdgcn_wmma_f32_16x16x32_bf16(false, ahi, false, bhi[kk],
                                                    (short)0, acc, false, false);
    }
    // D layout: VGPR r -> (M = r + 8*(lane>>4), N = lane&15)
    const int col   = nb + n0 + (lane & 15);
    const int rbase = r0 + t * 16 + ((lane >> 4) << 3);
    float* p = h1 + (size_t)rbase * HID_CH + col;
    if (full) {
#pragma unroll
      for (int r = 0; r < 8; ++r) p[(size_t)r * HID_CH] = acc[r];
    } else {
#pragma unroll
      for (int r = 0; r < 8; ++r)
        if (rbase + r < N_NODES) p[(size_t)r * HID_CH] = acc[r];
    }
  }
}

// ---------------------------------------------------------------------------
// GEMM2: h2[N,64] = helu[N,128] @ W2[128,64], helu given as bf16 hi/lo planes.
// Block: 256 thr = 8 waves, 64 rows x 64 cols. K = 128 = 4 x 32.
// A-operand staging uses GLOBAL_LOAD_ASYNC_TO_LDS_B128 when available.
// ---------------------------------------------------------------------------
__global__ __launch_bounds__(256)
void gemm2_h_w2(const __bf16* __restrict__ hhi, const __bf16* __restrict__ hlo,
                const float* __restrict__ W2, float* __restrict__ h2) {
  __shared__ __bf16 xs_hi[64][HID_CH + 8];
  __shared__ __bf16 xs_lo[64][HID_CH + 8];
  __shared__ __bf16 wt_hi[OUT_CH][HID_CH + 8];
  __shared__ __bf16 wt_lo[OUT_CH][HID_CH + 8];

  const int tid = threadIdx.x;
  const int r0  = blockIdx.x * 64;

  // Stage W2^T hi/lo (W2 is [128][64] row-major)
  for (int i = tid; i < HID_CH * OUT_CH; i += 256) {
    int k = i >> 6;
    int n = i & 63;
    float v = W2[i];
    __bf16 h = (__bf16)v;
    wt_hi[n][k] = h;
    wt_lo[n][k] = (__bf16)(v - (float)h);
  }

#if USE_ASYNC_LDS
  // Pure byte-copy staging: 64 rows x 256 B per plane = 1024 x 16B chunks.
  {
    const AS1 char* ghi = (const AS1 char*)hhi;
    const AS1 char* glo = (const AS1 char*)hlo;
    for (int c = tid; c < 1024; c += 256) {
      int rr = c >> 4;
      int cb = (c & 15) << 4;
      int row = r0 + rr;
      if (row < N_NODES) {
        __builtin_amdgcn_global_load_async_to_lds_b128(
            (AS1 v4i*)(ghi + (size_t)row * (HID_CH * 2) + cb),
            (AS3 v4i*)((AS3 char*)&xs_hi[rr][0] + cb), 0, 0);
        __builtin_amdgcn_global_load_async_to_lds_b128(
            (AS1 v4i*)(glo + (size_t)row * (HID_CH * 2) + cb),
            (AS3 v4i*)((AS3 char*)&xs_lo[rr][0] + cb), 0, 0);
      } else {
        float4 z = make_float4(0.f, 0.f, 0.f, 0.f);
        *(float4*)((char*)&xs_hi[rr][0] + cb) = z;
        *(float4*)((char*)&xs_lo[rr][0] + cb) = z;
      }
    }
    __builtin_amdgcn_s_wait_asynccnt(0);
  }
#else
  for (int i = tid; i < 64 * HID_CH; i += 256) {
    int r = i >> 7;
    int c = i & 127;
    __bf16 vh = (__bf16)0.0f, vl = (__bf16)0.0f;
    if (r0 + r < N_NODES) {
      size_t idx = (size_t)(r0 + r) * HID_CH + c;
      vh = hhi[idx];
      vl = hlo[idx];
    }
    xs_hi[r][c] = vh;
    xs_lo[r][c] = vl;
  }
#endif
  __syncthreads();

  const int wave = tid >> 5, lane = tid & 31;
  const int m  = lane & 15;
  const int kh = (lane >> 4) << 3;
  const int n0 = (wave & 3) << 4;
  const int tb = (wave >> 2) << 1;

  v16bf bhi[4], blo[4];
#pragma unroll
  for (int kk = 0; kk < 4; ++kk) {
    bhi[kk] = load_frag(&wt_hi[n0 + m][kk * 32 + kh]);
    blo[kk] = load_frag(&wt_lo[n0 + m][kk * 32 + kh]);
  }

  const bool full = (r0 + 64 <= N_NODES);

#pragma unroll
  for (int tt = 0; tt < 2; ++tt) {
    const int t = tb + tt;
    v8f acc = {};
#pragma unroll
    for (int kk = 0; kk < 4; ++kk) {
      v16bf ahi = load_frag(&xs_hi[t * 16 + m][kk * 32 + kh]);
      v16bf alo = load_frag(&xs_lo[t * 16 + m][kk * 32 + kh]);
      acc = __builtin_amdgcn_wmma_f32_16x16x32_bf16(false, ahi, false, blo[kk],
                                                    (short)0, acc, false, false);
      acc = __builtin_amdgcn_wmma_f32_16x16x32_bf16(false, alo, false, bhi[kk],
                                                    (short)0, acc, false, false);
      acc = __builtin_amdgcn_wmma_f32_16x16x32_bf16(false, ahi, false, bhi[kk],
                                                    (short)0, acc, false, false);
    }
    const int col   = n0 + (lane & 15);
    const int rbase = r0 + t * 16 + ((lane >> 4) << 3);
    float* p = h2 + (size_t)rbase * OUT_CH + col;
    if (full) {
#pragma unroll
      for (int r = 0; r < 8; ++r) p[(size_t)r * OUT_CH] = acc[r];
    } else {
#pragma unroll
      for (int r = 0; r < 8; ++r)
        if (rbase + r < N_NODES) p[(size_t)r * OUT_CH] = acc[r];
    }
  }
}

// ---------------------------------------------------------------------------
// Edge scatter: one wave per edge; edge indices via readfirstlane -> SMEM.
// fp32 hardware atomics resolve in the 192MB L2 (h/agg fully resident).
// ---------------------------------------------------------------------------
__global__ __launch_bounds__(256)
void scatter128(const long long* __restrict__ ei, const float* __restrict__ h,
                float* __restrict__ agg) {
  int e = __builtin_amdgcn_readfirstlane(blockIdx.x * 8 + (threadIdx.x >> 5));
  if (e >= N_EDGES) return;
  int lane = threadIdx.x & 31;
  long long s = ei[e];
  long long d = ei[(size_t)N_EDGES + e];
  float4 v = ((const float4*)(h + (size_t)s * HID_CH))[lane];
  float* ap = agg + (size_t)d * HID_CH + lane * 4;
  unsafeAtomicAdd(ap + 0, v.x);
  unsafeAtomicAdd(ap + 1, v.y);
  unsafeAtomicAdd(ap + 2, v.z);
  unsafeAtomicAdd(ap + 3, v.w);
}

__global__ __launch_bounds__(256)
void scatter64(const long long* __restrict__ ei, const float* __restrict__ h,
               float* __restrict__ agg) {
  int e = __builtin_amdgcn_readfirstlane(blockIdx.x * 8 + (threadIdx.x >> 5));
  if (e >= N_EDGES) return;
  int lane = threadIdx.x & 31;
  long long s = ei[e];
  long long d = ei[(size_t)N_EDGES + e];
  float2 v = ((const float2*)(h + (size_t)s * OUT_CH))[lane];
  float* ap = agg + (size_t)d * OUT_CH + lane * 2;
  unsafeAtomicAdd(ap + 0, v.x);
  unsafeAtomicAdd(ap + 1, v.y);
}

// bias + ELU, emit bf16 hi/lo planes so GEMM2 keeps near-fp32 precision
__global__ __launch_bounds__(256)
void bias_elu_split(const float* __restrict__ agg, const float* __restrict__ b1,
                    __bf16* __restrict__ hi_p, __bf16* __restrict__ lo_p) {
  size_t i = (size_t)blockIdx.x * 256 + threadIdx.x;   // grid sized exactly
  float v = agg[i] + b1[i & (HID_CH - 1)];
  v = (v > 0.0f) ? v : (__expf(v) - 1.0f);
  __bf16 h = (__bf16)v;
  hi_p[i] = h;
  lo_p[i] = (__bf16)(v - (float)h);
}

// Row softmax over 64 channels: one wave per row, wave32 shfl_xor reductions
__global__ __launch_bounds__(256)
void softmax64(const float* __restrict__ agg2, const float* __restrict__ b2,
               float* __restrict__ out) {
  int row = blockIdx.x * 8 + (threadIdx.x >> 5);
  if (row >= N_NODES) return;
  int lane = threadIdx.x & 31;
  float2 v  = ((const float2*)(agg2 + (size_t)row * OUT_CH))[lane];
  float2 bb = ((const float2*)b2)[lane];
  v.x += bb.x; v.y += bb.y;
  float m = fmaxf(v.x, v.y);
#pragma unroll
  for (int o = 16; o > 0; o >>= 1) m = fmaxf(m, __shfl_xor(m, o, 32));
  float ex = __expf(v.x - m), ey = __expf(v.y - m);
  float s = ex + ey;
#pragma unroll
  for (int o = 16; o > 0; o >>= 1) s += __shfl_xor(s, o, 32);
  float inv = 1.0f / s;
  float2 r; r.x = ex * inv; r.y = ey * inv;
  ((float2*)(out + (size_t)row * OUT_CH))[lane] = r;
}

// ---------------------------------------------------------------------------
// Workspace (102.4 MB):
//   buf0 [0, 51.2M):      h1 -> later helu_hi (lower 25.6M) + helu_lo (upper)
//   buf1 [51.2M, 102.4M): agg1 -> later h2 (lower 25.6M) + agg2 (upper 25.6M)
// ---------------------------------------------------------------------------
extern "C" void kernel_launch(void* const* d_in, const int* in_sizes, int n_in,
                              void* d_out, int out_size, void* d_ws, size_t ws_size,
                              hipStream_t stream) {
  const float*     x  = (const float*)d_in[0];
  const long long* ei = (const long long*)d_in[1];
  const float*     W1 = (const float*)d_in[2];
  const float*     b1 = (const float*)d_in[3];
  const float*     W2 = (const float*)d_in[4];
  const float*     b2 = (const float*)d_in[5];
  float* out = (float*)d_out;

  char* ws = (char*)d_ws;
  const size_t S1 = (size_t)N_NODES * HID_CH * sizeof(float);   // 51.2 MB
  float*  h1   = (float*)(ws);
  float*  agg1 = (float*)(ws + S1);
  __bf16* hhi  = (__bf16*)(ws);                 // after scatter1
  __bf16* hlo  = (__bf16*)(ws + S1 / 2);
  float*  h2   = (float*)(ws + S1);             // after bias_elu
  float*  agg2 = (float*)(ws + S1 + S1 / 2);

  const int row_blocks = (N_NODES + 63) / 64;   // 1563

  (void)hipMemsetAsync(agg1, 0, S1, stream);
  gemm1_x_w1<<<dim3(row_blocks, 2), 256, 0, stream>>>(x, W1, h1);
  scatter128<<<N_EDGES / 8, 256, 0, stream>>>(ei, h1, agg1);
  bias_elu_split<<<(N_NODES * HID_CH) / 256, 256, 0, stream>>>(agg1, b1, hhi, hlo);
  (void)hipMemsetAsync(agg2, 0, (size_t)N_NODES * OUT_CH * sizeof(float), stream);
  gemm2_h_w2<<<row_blocks, 256, 0, stream>>>(hhi, hlo, W2, h2);
  scatter64<<<N_EDGES / 8, 256, 0, stream>>>(ei, h2, agg2);
  softmax64<<<(N_NODES + 7) / 8, 256, 0, stream>>>(agg2, b2, out);
}
